// RelationGraph_13134009991572
// MI455X (gfx1250) — matrix operand order repaired and checked by reading, not compiled, not run on hardware
//
#include <hip/hip_runtime.h>

typedef __attribute__((ext_vector_type(8)))  __bf16 v8bf;
typedef __attribute__((ext_vector_type(16))) __bf16 v16bf;
typedef __attribute__((ext_vector_type(8)))  float  v8f;
typedef __attribute__((ext_vector_type(4)))  float  v4f;

static constexpr int kB = 32, kR = 5, kN = 1024, kD = 256, kH = 256;
static constexpr int TN = 32;                 // n-rows per workgroup
static constexpr int A_STRIDE_H = kD + 8;     // padded bf16 row stride (halves) -> kills bank conflicts

static __device__ __forceinline__ unsigned short f32_bf16(float f) {
  union { float f; unsigned u; } v; v.f = f;
  unsigned r = v.u + 0x7FFFu + ((v.u >> 16) & 1u);   // round-to-nearest-even
  return (unsigned short)(r >> 16);
}

// ---- tiny pre-pass: fp32 weights -> bf16 in workspace --------------------
__global__ void __launch_bounds__(256)
cvt_kernel(const float* __restrict__ src, unsigned short* __restrict__ dst, int n4) {
  int i = blockIdx.x * 256 + threadIdx.x;
  if (i >= n4) return;
  v4f v = reinterpret_cast<const v4f*>(src)[i];
  uint2 p;
  p.x = (unsigned)f32_bf16(v.x) | ((unsigned)f32_bf16(v.y) << 16);
  p.y = (unsigned)f32_bf16(v.z) | ((unsigned)f32_bf16(v.w) << 16);
  reinterpret_cast<uint2*>(dst)[i] = p;
}

// ---- fused: adj row-sums + 6 accumulated bf16 WMMA GEMMs -----------------
__global__ void __launch_bounds__(256)
rgcn_fused_kernel(const float* __restrict__ nodes,
                  const float* __restrict__ adj,
                  const unsigned short* __restrict__ wbf,   // [H*D] w0, then [R*H*D] w_r (bf16 bits)
                  float* __restrict__ out) {
  __shared__ __align__(16) float        nodesLds[TN * kD];            // 32 KB fp32 tile
  __shared__ __align__(16) unsigned int ALds[TN * A_STRIDE_H / 2];    // padded bf16 tile
  __shared__ float invc[kR * TN];

  const int tid  = threadIdx.x;
  const int lane = tid & 31;
  const int wave = tid >> 5;
  const int b     = blockIdx.x >> 5;          // 32 n-tiles per batch
  const int nBase = (blockIdx.x & 31) * TN;

  // Phase 1: invc[r][row] = 1 / rowsum(adj[b,r,nBase+row,:])  (streamed once, NT)
  for (int idx = wave; idx < kR * TN; idx += 8) {
    const int r   = idx >> 5;
    const int row = idx & 31;
    const v4f* arow4 = reinterpret_cast<const v4f*>(
        adj + (((size_t)b * kR + r) * kN + (nBase + row)) * (size_t)kN);
    float s = 0.f;
    #pragma unroll
    for (int j = 0; j < 8; ++j) {
      v4f v = __builtin_nontemporal_load(&arow4[lane + 32 * j]);
      s += v.x + v.y + v.z + v.w;
    }
    #pragma unroll
    for (int off = 16; off > 0; off >>= 1) s += __shfl_xor(s, off, 32);
    if (lane == 0) invc[idx] = (s == 0.f) ? 1.f : 1.f / s;
  }

  // Phase 2: nodes tile -> LDS (tile is one contiguous 8192-float block)
  {
    const v4f* src = reinterpret_cast<const v4f*>(nodes + ((size_t)b * kN + nBase) * kD);
    v4f* dst = reinterpret_cast<v4f*>(nodesLds);
    #pragma unroll
    for (int j = 0; j < 8; ++j) dst[tid + 256 * j] = src[tid + 256 * j];
  }

  // wave -> output subtile mapping: 2 (n) x 4 (h) waves, 16x64 each = 4 accs
  const int nSub = wave >> 2;
  const int hGrp = wave & 3;
  const int l16  = lane & 15;
  const int kSel = lane >> 4;
  const __bf16* aBase = reinterpret_cast<const __bf16*>(ALds)
                      + (size_t)(nSub * 16 + l16) * A_STRIDE_H;

  v8f acc[4] = {};

  for (int m = 0; m < 6; ++m) {
    __syncthreads();   // previous k-loop done reading ALds (and phases 1/2 on first iter)
    // build row-scaled bf16 A tile: scale = 1 (w0) or invc[m-1][row]
    #pragma unroll
    for (int j = 0; j < 16; ++j) {
      const int p   = tid + 256 * j;     // pair index over 32x256 halves
      const int row = p >> 7;
      const int cp  = p & 127;           // pair-column within row
      const float sc = (m == 0) ? 1.f : invc[(m - 1) * TN + row];
      const float2 v = reinterpret_cast<const float2*>(nodesLds)[p];
      ALds[row * (A_STRIDE_H / 2) + cp] =
          (unsigned)f32_bf16(v.x * sc) | ((unsigned)f32_bf16(v.y * sc) << 16);
    }
    __syncthreads();

    const __bf16* w = reinterpret_cast<const __bf16*>(wbf) + (size_t)m * kH * kD;
    #pragma unroll
    for (int kt = 0; kt < 8; ++kt) {
      // A frag: contiguous 16 halves at kt*32 + kSel*16 (matches B)
      const int kb = kt * 32 + kSel * 16;
      v8bf a0 = *reinterpret_cast<const v8bf*>(aBase + kb);
      v8bf a1 = *reinterpret_cast<const v8bf*>(aBase + kb + 8);
      v16bf a = __builtin_shufflevector(a0, a1, 0,1,2,3,4,5,6,7,8,9,10,11,12,13,14,15);
      #pragma unroll
      for (int ht = 0; ht < 4; ++ht) {
        // B frag: lane = column h, contiguous 16 halves at kt*32 + kSel*16 (B[k][n] = W[h][k] -> A@W^T)
        const int h = hGrp * 64 + ht * 16 + l16;
        v16bf bb = *reinterpret_cast<const v16bf*>(w + (size_t)h * kD + kt * 32 + kSel * 16);
        acc[ht] = __builtin_amdgcn_wmma_f32_16x16x32_bf16(
            false, a, false, bb, (short)0, acc[ht], false, false);
      }
    }
  }

  // store: C/D layout -> lane<16: M=j, lane>=16: M=8+j, N=lane%16
  const int rowOutBase = nBase + nSub * 16 + kSel * 8;
  #pragma unroll
  for (int ht = 0; ht < 4; ++ht) {
    const int h = hGrp * 64 + ht * 16 + l16;
    #pragma unroll
    for (int j = 0; j < 8; ++j) {
      out[((size_t)b * kN + (rowOutBase + j)) * kH + h] = acc[ht][j];
    }
  }
}

extern "C" void kernel_launch(void* const* d_in, const int* in_sizes, int n_in,
                              void* d_out, int out_size, void* d_ws, size_t ws_size,
                              hipStream_t stream) {
  const float* nodes = (const float*)d_in[0];   // (B,N,D)
  const float* adj   = (const float*)d_in[1];   // (B,R,N,N)
  const float* w0    = (const float*)d_in[2];   // (H,D)
  const float* wr    = (const float*)d_in[3];   // (R,H,D)
  unsigned short* wbf = (unsigned short*)d_ws;  // (1+R)*H*D bf16 = 786 KB scratch

  const int n4_w0 = kH * kD / 4;
  const int n4_wr = kR * kH * kD / 4;
  cvt_kernel<<<(n4_w0 + 255) / 256, 256, 0, stream>>>(w0, wbf, n4_w0);
  cvt_kernel<<<(n4_wr + 255) / 256, 256, 0, stream>>>(wr, wbf + kH * kD, n4_wr);

  rgcn_fused_kernel<<<kB * (kN / TN), 256, 0, stream>>>(nodes, adj, wbf, (float*)d_out);
}
